// TLSTM_65807488909847
// MI455X (gfx1250) — matrix operand order repaired and checked by reading, not compiled, not run on hardware
//
#include <hip/hip_runtime.h>
#include <hip/hip_bf16.h>

#define B_ 128
#define S_ 512
#define I_ 256
#define H_ 512

typedef __attribute__((ext_vector_type(16))) __bf16 v16bf;
typedef __attribute__((ext_vector_type(4)))  __bf16 v4bf;
typedef __attribute__((ext_vector_type(8)))  float  v8f;

// GCC-style vector types (valid for host semantic checking too).
typedef int   v4i_g __attribute__((__vector_size__(16)));
typedef float v4f_g __attribute__((__vector_size__(16)));
typedef __attribute__((address_space(1))) v4i_g* as1_v4i_ptr;
typedef __attribute__((address_space(3))) v4i_g* as3_v4i_ptr;
// Explicit global-AS fragment pointer: keeps laundered weight loads as
// global_load_b128 (LOADcnt only) instead of flat_load (LOADcnt+DScnt).
typedef const __attribute__((address_space(1))) v16bf* as1_v16bf_cptr;

#if defined(__has_builtin)
#  if __has_builtin(__builtin_amdgcn_global_load_async_to_lds_b128)
#    define HAVE_ASYNC_LDS 1
#  else
#    define HAVE_ASYNC_LDS 0
#  endif
#  if __has_builtin(__builtin_amdgcn_s_wait_asynccnt)
#    define HAVE_WAIT_ASYNC 1
#  else
#    define HAVE_WAIT_ASYNC 0
#  endif
#else
#  define HAVE_ASYNC_LDS 0
#  define HAVE_WAIT_ASYNC 0
#endif

__device__ __forceinline__ float sigmoidf_(float x) {
    return 1.0f / (1.0f + __expf(-x));
}

// ---------------------------------------------------------------------------
// One-time pre-pass: f32 (K x N row-major) weights -> bf16 B-fragment order.
// Per 32x16 (KxN) tile: 32 lanes x 16 contiguous bf16, matching
// V_WMMA_*_16X16X32_BF16 B layout (lane L: n = L&15, K = (e>>3)*16 + (L>>4)*8 + (e&7)).
// ---------------------------------------------------------------------------
__global__ void swizzle_w(const float* __restrict__ src, __bf16* __restrict__ dst,
                          int K, int N) {
    int id = blockIdx.x * blockDim.x + threadIdx.x;
    if (id >= K * N) return;
    int e    = id & 15;
    int L    = (id >> 4) & 31;
    int tile = id >> 9;
    int ntn  = N >> 4;
    int nt   = tile % ntn;
    int kt   = tile / ntn;
    int k = kt * 32 + ((e >> 3) << 4) + ((L >> 4) << 3) + (e & 7);
    int n = nt * 16 + (L & 15);
    dst[id] = (__bf16)src[k * N + n];
}

// B fragment at (per-lane base) + compile-time byte offset. The constant
// offset folds into the global_load_b128 instruction offset field (24-bit).
__device__ __forceinline__ v16bf load_b_off(uintptr_t base, int byteOff) {
    return *(as1_v16bf_cptr)(base + (size_t)byteOff);
}

#define WMMA_BF16(A, Bf, C) \
    __builtin_amdgcn_wmma_f32_16x16x32_bf16(false, (A), false, (Bf), (short)0, (C), false, false)

// Byte strides in the swizzled weight buffers:
//   one 32x16 tile      = 32 lanes * 16 bf16 * 2B = 1024 B
//   one k-row of W/U    = 128 tiles               = 131072 B
//   one gate block      = 32 tiles                = 32768 B
//   one k-row of W_d    = 32 tiles                = 32768 B
#define TILE_B   1024
#define KROW_B   131072
#define GATE_B   32768
#define KROWD_B  32768

// ---------------------------------------------------------------------------
// Persistent T-LSTM: one workgroup (32 waves) owns one 16-row batch tile and
// runs the full 512-step recurrence with LDS-resident state.
// Wave w = H col-tile w (16 columns). Grid = 8 blocks, 1024 threads.
// ---------------------------------------------------------------------------
__global__ __launch_bounds__(1024) void tlstm_persistent(
    const float* __restrict__ x,    // (B,S,I) f32
    const float* __restrict__ dtv,  // (B,S,1) f32
    const float* __restrict__ b,    // (4H) f32
    const float* __restrict__ bd,   // (H)  f32
    const __bf16* __restrict__ Wf,  // swizzled (I x 4H)
    const __bf16* __restrict__ Uf,  // swizzled (H x 4H)
    const __bf16* __restrict__ Wdf, // swizzled (H x H)
    float* __restrict__ out)        // (B,S,H) ++ h_t(B,H) ++ c_t(B,H)
{
    __shared__ __bf16 xS[8  * 32 * 16];  // x_t tile, A-fragment order (8 KB)
    __shared__ __bf16 hA[16 * 32 * 16];  // h state,  A-fragment order (16 KB)
    __shared__ __bf16 cA[16 * 32 * 16];  // c state,  A-fragment order (16 KB)
#if HAVE_ASYNC_LDS
    __shared__ float xRaw[16 * I_];      // raw x_t tile (16 KB), DMA target
#endif

    const int tid     = threadIdx.x;
    const int lane    = tid & 31;
    const int colTile = tid >> 5;        // 0..31
    const int rowTile = blockIdx.x;      // 0..7
    const int rowbase = rowTile * 16;
    const int colbase = colTile * 16;
    const int n       = lane & 15;
    const int khalf   = lane >> 4;

    // Zero LDS state (h0 = c0 = 0).
    for (int i = tid; i < 16 * 32 * 16; i += 1024) {
        hA[i] = (__bf16)0.0f;
        cA[i] = (__bf16)0.0f;
    }
    __syncthreads();

    // Scalar biases (added in the pointwise phase, so accumulators can be
    // zero-initialized -> WMMA C starts as inline constant, nothing to spill).
    const float bi0 = b[0 * H_ + colbase + n];
    const float bf0 = b[1 * H_ + colbase + n];
    const float bo0 = b[2 * H_ + colbase + n];
    const float bc0 = b[3 * H_ + colbase + n];
    const float bs0 = bd[colbase + n];

    // f32 cell state never leaves registers: lane owns its 8 C-layout elems.
    float cPrev[8];
    #pragma unroll
    for (int v = 0; v < 8; ++v) cPrev[v] = 0.0f;

    const int ktile_w = colTile >> 1;        // k-tile our columns fall in
    const int kin     = (colTile & 1) * 16;  // k offset within that k-tile
    const float Ec    = 2.718281828459045f;

    // Per-lane byte offset of this wave's column tile + lane within any
    // swizzled weight buffer (single small loop-invariant scalar).
    const int laneByte = colTile * TILE_B + lane * 32;

    for (int t = 0; t < S_; ++t) {
        // Launder the (loop-invariant) weight pointers so the compiler cannot
        // hoist all 112 B-fragment loads out of the t-loop and spill them to
        // scratch; loads stay as per-step global_load_b128 from L2.
        uintptr_t wfp = (uintptr_t)Wf, ufp = (uintptr_t)Uf, wdp = (uintptr_t)Wdf;
        asm volatile("" : "+s"(wfp), "+s"(ufp), "+s"(wdp));
        const uintptr_t wB = wfp + laneByte;   // 3 adds per step, total
        const uintptr_t uB = ufp + laneByte;
        const uintptr_t dB = wdp + laneByte;

        // ---- [A] stage x_t (16 x 256 f32) into fragment-ordered bf16 LDS ----
#if HAVE_ASYNC_LDS
        {
            int m   = tid >> 6;        // row 0..15
            int c16 = tid & 63;        // 16-byte chunk within the row
            const float* g = x + ((size_t)(rowbase + m) * S_ + t) * I_ + c16 * 4;
            __builtin_amdgcn_global_load_async_to_lds_b128(
                (as1_v4i_ptr)(uintptr_t)g,
                (as3_v4i_ptr)(uintptr_t)(&xRaw[tid * 4]),
                0, 0);
        }
#if HAVE_WAIT_ASYNC
        __builtin_amdgcn_s_wait_asynccnt(0);
#else
        asm volatile("s_wait_asynccnt 0x0" ::: "memory");
#endif
        __syncthreads();
#endif
        {
            int id = tid * 4;                       // 4 frag elems per thread
            int e0 = id & 15, L = (id >> 4) & 31, kt = (id >> 9) & 7;
            int m = L & 15, koff = (L >> 4) << 3;
            int k = kt * 32 + ((e0 >> 3) << 4) + koff + (e0 & 7);
            v4bf vv;
#if HAVE_ASYNC_LDS
            const float* p = &xRaw[m * I_ + k];
            vv[0] = (__bf16)p[0]; vv[1] = (__bf16)p[1];
            vv[2] = (__bf16)p[2]; vv[3] = (__bf16)p[3];
#else
            const v4f_g xv = __builtin_nontemporal_load(
                (const v4f_g*)(x + ((size_t)(rowbase + m) * S_ + t) * I_ + k));
            vv[0] = (__bf16)xv[0]; vv[1] = (__bf16)xv[1];
            vv[2] = (__bf16)xv[2]; vv[3] = (__bf16)xv[3];
#endif
            *(v4bf*)(xS + id) = vv;
        }
        __syncthreads();

        // ---- [B] GEMM phase: 112 WMMAs per wave, zero-init accumulators ----
        v8f accI = {0, 0, 0, 0, 0, 0, 0, 0};
        v8f accF = {0, 0, 0, 0, 0, 0, 0, 0};
        v8f accO = {0, 0, 0, 0, 0, 0, 0, 0};
        v8f accC = {0, 0, 0, 0, 0, 0, 0, 0};
        v8f accS = {0, 0, 0, 0, 0, 0, 0, 0};

        #pragma unroll
        for (int kt = 0; kt < I_ / 32; ++kt) {         // z += x_t @ W
            v16bf a = *(const v16bf*)(xS + kt * 512 + lane * 16);
            accI = WMMA_BF16(a, load_b_off(wB, kt * KROW_B + 0 * GATE_B), accI);
            accF = WMMA_BF16(a, load_b_off(wB, kt * KROW_B + 1 * GATE_B), accF);
            accO = WMMA_BF16(a, load_b_off(wB, kt * KROW_B + 2 * GATE_B), accO);
            accC = WMMA_BF16(a, load_b_off(wB, kt * KROW_B + 3 * GATE_B), accC);
        }
        #pragma unroll
        for (int kt = 0; kt < H_ / 32; ++kt) {         // z += h @ U
            v16bf a = *(const v16bf*)(hA + kt * 512 + lane * 16);
            accI = WMMA_BF16(a, load_b_off(uB, kt * KROW_B + 0 * GATE_B), accI);
            accF = WMMA_BF16(a, load_b_off(uB, kt * KROW_B + 1 * GATE_B), accF);
            accO = WMMA_BF16(a, load_b_off(uB, kt * KROW_B + 2 * GATE_B), accO);
            accC = WMMA_BF16(a, load_b_off(uB, kt * KROW_B + 3 * GATE_B), accC);
        }
        #pragma unroll
        for (int kt = 0; kt < H_ / 32; ++kt) {         // c_s_pre = c @ W_d
            v16bf a = *(const v16bf*)(cA + kt * 512 + lane * 16);
            accS = WMMA_BF16(a, load_b_off(dB, kt * KROWD_B), accS);
        }
        __syncthreads();   // all reads of hA/cA complete before state update

        // ---- [C] pointwise T-LSTM update (biases added here) ----
        #pragma unroll
        for (int v = 0; v < 8; ++v) {
            int m    = v + khalf * 8;       // C/D layout row
            int brow = rowbase + m;
            int col  = colbase + n;

            float d    = dtv[(size_t)brow * S_ + t];
            float gdec = 1.0f / logf(Ec + d);

            float cs   = tanhf(accS[v] + bs0);
            float cadj = cPrev[v] - cs + cs * gdec;
            float ig   = sigmoidf_(accI[v] + bi0);
            float fg   = sigmoidf_(accF[v] + bf0);
            float og   = sigmoidf_(accO[v] + bo0);
            float ct   = tanhf(accC[v] + bc0);
            float cnew = fg * cadj + ig * ct;
            float hnew = og * tanhf(cnew);
            cPrev[v] = cnew;

            __builtin_nontemporal_store(hnew,
                out + ((size_t)brow * S_ + t) * H_ + col);

            // scatter new state into A-fragment order
            int kq   = kin + n;                       // 0..31 within k-tile
            int Lf   = m + ((n >= 8) ? 16 : 0);
            int e    = ((kq >= 16) ? 8 : 0) + (n & 7);
            int fidx = ktile_w * 512 + Lf * 16 + e;
            hA[fidx] = (__bf16)hnew;
            cA[fidx] = (__bf16)cnew;

            if (t == S_ - 1) {
                out[(size_t)B_ * S_ * H_ + (size_t)brow * H_ + col] = hnew;
                out[(size_t)B_ * S_ * H_ + (size_t)B_ * H_ +
                    (size_t)brow * H_ + col] = cnew;
            }
        }
        __syncthreads();   // state visible before next step's GEMM reads
    }
}

extern "C" void kernel_launch(void* const* d_in, const int* in_sizes, int n_in,
                              void* d_out, int out_size, void* d_ws, size_t ws_size,
                              hipStream_t stream) {
    (void)in_sizes; (void)n_in; (void)out_size; (void)ws_size;
    const float* x  = (const float*)d_in[0];
    const float* dt = (const float*)d_in[1];
    const float* W  = (const float*)d_in[2];
    const float* U  = (const float*)d_in[3];
    const float* b  = (const float*)d_in[4];
    const float* Wd = (const float*)d_in[5];
    const float* bd = (const float*)d_in[6];
    float* out = (float*)d_out;

    // Workspace: swizzled bf16 weights only (~3.5 MB, L2-resident).
    char* ws = (char*)d_ws;
    __bf16* Wf  = (__bf16*)ws;  ws += (size_t)I_ * 4 * H_ * sizeof(__bf16);
    __bf16* Uf  = (__bf16*)ws;  ws += (size_t)H_ * 4 * H_ * sizeof(__bf16);
    __bf16* Wdf = (__bf16*)ws;  ws += (size_t)H_ * H_ * sizeof(__bf16);

    swizzle_w<<<(I_ * 4 * H_ + 255) / 256, 256, 0, stream>>>(W,  Wf,  I_, 4 * H_);
    swizzle_w<<<(H_ * 4 * H_ + 255) / 256, 256, 0, stream>>>(U,  Uf,  H_, 4 * H_);
    swizzle_w<<<(H_ * H_     + 255) / 256, 256, 0, stream>>>(Wd, Wdf, H_, H_);

    // Entire 512-step recurrence in one persistent kernel:
    // 8 independent batch row-tiles -> 8 workgroups of 32 waves.
    tlstm_persistent<<<dim3(8), 1024, 0, stream>>>(x, dt, b, bd, Wf, Uf, Wdf, out);
}
